// MotionFormer_68805376082090
// MI455X (gfx1250) — compile-verified, matrix operand order checked by reading, not run on hardware
//
#include <hip/hip_runtime.h>
#include <hip/hip_bf16.h>
#include <math.h>

// ---------------------------------------------------------------------------
// MotionFormer forward for MI455X (gfx1250), wave32 + WMMA bf16.
// All big GEMMs use v_wmma_f32_16x16x32_bf16 (f32 accumulate).
// ---------------------------------------------------------------------------

#define N_  2
#define A_  256
#define M_  300
#define K_  6
#define D_  256
#define H_  8
#define T_  12
#define L_  3
#define P_  4
#define FF_ 512
#define HB_ 200
#define WB_ 200
#define LQ_ (A_ * K_)          // 1536
#define BEV_ (HB_ * WB_)       // 40000

typedef __attribute__((ext_vector_type(16))) __bf16 v16bf;
typedef __attribute__((ext_vector_type(8)))  float  v8f;

__device__ __forceinline__ __bf16 f2bf(float f) {
  union { float f; unsigned int u; } c; c.f = f;
  unsigned int u = c.u;
  u += 0x7FFFu + ((u >> 16) & 1u);       // round to nearest even
  union { unsigned short s; __bf16 b; } r;
  r.s = (unsigned short)(u >> 16);
  return r.b;
}

// K-index inside a 16x32 (16-bit) A/B WMMA fragment for element e (0..15)
// per CDNA5 ISA 7.12.2: VGPR v in 0..3 -> K = 2v,2v+1 (+8 for lanes 16..31),
// VGPR 4..7 -> +16.  Groups of 8 elements are contiguous -> ds_load_b128.
__device__ __forceinline__ int kmap(int e, int half) {
  return ((e >> 3) << 4) + (((e >> 1) & 3) << 1) + (e & 1) + (half << 3);
}

// ---- DPP16 cross-lane reduction within a 16-lane half (VALU only, no LDS).
// quad_perm(1,0,3,2)=0xB1, quad_perm(2,3,0,1)=0x4E,
// row_half_mirror=0x141, row_mirror=0x140 -- all stay inside 16 lanes.
template <int CTRL>
__device__ __forceinline__ float dppf(float x) {
  return __builtin_bit_cast(
      float, __builtin_amdgcn_update_dpp(0, __builtin_bit_cast(int, x), CTRL,
                                         0xF, 0xF, true));
}
__device__ __forceinline__ float redmax16(float m) {
  m = fmaxf(m, dppf<0xB1>(m));
  m = fmaxf(m, dppf<0x4E>(m));
  m = fmaxf(m, dppf<0x141>(m));
  m = fmaxf(m, dppf<0x140>(m));
  return m;
}
__device__ __forceinline__ float redsum16(float s) {
  s += dppf<0xB1>(s);
  s += dppf<0x4E>(s);
  s += dppf<0x141>(s);
  s += dppf<0x140>(s);
  return s;
}

// ---------------------------------------------------------------------------
// GEMM:  C[M,N] = act(A[M,K] @ W[K,N] + bias[N]).  K % 64 == 0.
// Block = 128 threads (4 waves). Block tile 128x64, each wave 32x64.
// K staged 64 at a time (two 32-wide sub-steps) -> 16 WMMAs per barrier pair.
// ---------------------------------------------------------------------------
template <int ACT>
__global__ void __launch_bounds__(128)
gemm_bias_act(const float* __restrict__ A, const float* __restrict__ W,
              const float* __restrict__ bias, float* __restrict__ C,
              int M, int N, int K)
{
  __shared__ __bf16 As[2][128][32];
  __shared__ __bf16 Ws[2][32][64];
  const int tid  = threadIdx.x;
  const int wave = tid >> 5;
  const int lane = tid & 31;
  const int half = lane >> 4;
  const int idx  = lane & 15;
  const int rowBase = blockIdx.y * 128;
  const int colBase = blockIdx.x * 64;

  v8f acc[2][4];
#pragma unroll
  for (int r = 0; r < 2; ++r)
#pragma unroll
    for (int j = 0; j < 4; ++j) acc[r][j] = (v8f)0.0f;

  for (int k0 = 0; k0 < K; k0 += 64) {
    // stage A: 128 rows x 64 k (two 32-wide sub-tiles)
    for (int t = tid; t < 128 * 64; t += 128) {
      int r = t >> 6, c = t & 63;
      int gr = rowBase + r;
      float v = (gr < M) ? A[(size_t)gr * K + (k0 + c)] : 0.0f;
      As[c >> 5][r][c & 31] = f2bf(v);
    }
    // stage W: 64 k x 64 cols
    for (int t = tid; t < 64 * 64; t += 128) {
      int r = t >> 6, c = t & 63;
      int gc = colBase + c;
      float v = (gc < N) ? W[(size_t)(k0 + r) * N + gc] : 0.0f;
      Ws[r >> 5][r & 31][c] = f2bf(v);
    }
    __syncthreads();

    // prefetch next K-tile while this one computes (global_prefetch_b8)
    if (k0 + 64 < K) {
      int pr = rowBase + tid;
      if (pr < M) __builtin_prefetch(&A[(size_t)pr * K + (k0 + 64)], 0, 1);
      if (tid < 64) {
        int pc = colBase + ((tid & 1) * 32);
        if (pc < N)
          __builtin_prefetch(&W[(size_t)(k0 + 64 + (tid >> 1)) * N + pc], 0, 1);
      }
    }

#pragma unroll
    for (int s = 0; s < 2; ++s) {
      v16bf af0, af1;
      const int ar = wave * 32 + idx;
#pragma unroll
      for (int e = 0; e < 16; ++e) {
        const int kk = kmap(e, half);
        af0[e] = As[s][ar][kk];
        af1[e] = As[s][ar + 16][kk];
      }
#pragma unroll
      for (int j = 0; j < 4; ++j) {
        v16bf bfr;
        const int bcol = j * 16 + idx;
#pragma unroll
        for (int e = 0; e < 16; ++e) bfr[e] = Ws[s][kmap(e, half)][bcol];
        acc[0][j] = __builtin_amdgcn_wmma_f32_16x16x32_bf16(
            false, af0, false, bfr, (short)0, acc[0][j], false, false);
        acc[1][j] = __builtin_amdgcn_wmma_f32_16x16x32_bf16(
            false, af1, false, bfr, (short)0, acc[1][j], false, false);
      }
    }
    __syncthreads();
  }

#pragma unroll
  for (int j = 0; j < 4; ++j) {
    const int col = colBase + j * 16 + idx;
    if (col < N) {
      const float bv = bias ? bias[col] : 0.0f;
#pragma unroll
      for (int r = 0; r < 2; ++r) {
#pragma unroll
        for (int g = 0; g < 8; ++g) {
          const int row = rowBase + wave * 32 + r * 16 + g + half * 8;
          if (row < M) {
            float v = acc[r][j][g] + bv;
            if (ACT == 1) v = v > 0.0f ? v : 0.0f;
            C[(size_t)row * N + col] = v;
          }
        }
      }
    }
  }
}

// ---------------------------------------------------------------------------
// Flash attention (per head, hd = 32): O = softmax(Q K^T / sqrt(hd)) V
// Q/K/V/O layout: [NB, L, D] with head h occupying cols [h*32, h*32+32).
// Block = 128 threads (4 waves), 64 queries per block, 64-key chunks.
// WMMA for both Q*K^T and P*V; DPP16 for the softmax row reductions.
// ---------------------------------------------------------------------------
__global__ void __launch_bounds__(128)
flash_attn(const float* __restrict__ Q, const float* __restrict__ Kv,
           const float* __restrict__ Vv, float* __restrict__ O,
           int LQ, int LK, float scale)
{
  __shared__ __bf16 Ks[64][32];
  __shared__ __bf16 Vs[64][32];
  __shared__ __bf16 Ps[4][16][64];

  const int tid  = threadIdx.x;
  const int wave = tid >> 5;
  const int lane = tid & 31;
  const int half = lane >> 4;
  const int idx  = lane & 15;
  const int h    = blockIdx.y;
  const int nb   = blockIdx.z;
  const int qbase  = blockIdx.x * 64;
  const int colOff = h * 32;

  // Q fragment (16 queries x 32 dims per wave)
  v16bf qf;
  {
    const int q = qbase + wave * 16 + idx;
    const float* qrow = Q + ((size_t)nb * LQ + (q < LQ ? q : 0)) * D_ + colOff;
#pragma unroll
    for (int e = 0; e < 16; ++e)
      qf[e] = f2bf((q < LQ) ? qrow[kmap(e, half)] : 0.0f);
  }

  float rm[8], rl[8];
#pragma unroll
  for (int g = 0; g < 8; ++g) { rm[g] = -1e30f; rl[g] = 0.0f; }
  v8f oa = (v8f)0.0f, ob = (v8f)0.0f;

  for (int kc = 0; kc < LK; kc += 64) {
    for (int t = tid; t < 64 * 32; t += 128) {
      int r = t >> 5, c = t & 31;
      int gk = kc + r;
      float kvv = 0.0f, vvv = 0.0f;
      if (gk < LK) {
        size_t off = ((size_t)nb * LK + gk) * D_ + colOff + c;
        kvv = Kv[off]; vvv = Vv[off];
      }
      Ks[r][c] = f2bf(kvv);
      Vs[r][c] = f2bf(vvv);
    }
    __syncthreads();

    // S = Q @ K^T for this 64-key chunk (4 WMMAs)
    v8f s[4];
#pragma unroll
    for (int j = 0; j < 4; ++j) {
      v16bf bfr;
      const int kl = j * 16 + idx;
#pragma unroll
      for (int e = 0; e < 16; ++e) bfr[e] = Ks[kl][kmap(e, half)];
      v8f z = (v8f)0.0f;
      s[j] = __builtin_amdgcn_wmma_f32_16x16x32_bf16(
          false, qf, false, bfr, (short)0, z, false, false);
    }

    // scale + key-range mask
#pragma unroll
    for (int j = 0; j < 4; ++j) {
      const bool ok = (kc + j * 16 + idx) < LK;
#pragma unroll
      for (int g = 0; g < 8; ++g)
        s[j][g] = ok ? s[j][g] * scale : -1e30f;
    }

    // online softmax: per-row max & sum via DPP16 (rows: g + 8*half;
    // 64 cols spread over 16 lanes x 4 accumulators)
    float corr[8];
#pragma unroll
    for (int g = 0; g < 8; ++g) {
      float m = fmaxf(fmaxf(s[0][g], s[1][g]), fmaxf(s[2][g], s[3][g]));
      m = redmax16(m);
      const float nm = fmaxf(rm[g], m);
      corr[g] = __expf(rm[g] - nm);
      rm[g] = nm;
    }
#pragma unroll
    for (int g = 0; g < 8; ++g) {
      float sum = 0.0f;
#pragma unroll
      for (int j = 0; j < 4; ++j) {
        const float p = __expf(s[j][g] - rm[g]);
        s[j][g] = p;
        sum += p;
      }
      sum = redsum16(sum);
      rl[g] = rl[g] * corr[g] + sum;
      oa[g] *= corr[g];
      ob[g] *= corr[g];
    }

    // stage P (C-layout) into LDS, reload as A-fragments (wave-private)
#pragma unroll
    for (int j = 0; j < 4; ++j)
#pragma unroll
      for (int g = 0; g < 8; ++g)
        Ps[wave][g + half * 8][j * 16 + idx] = f2bf(s[j][g]);

    // O += P @ V  (2 key-halves x 2 d-col-blocks = 4 WMMAs)
#pragma unroll
    for (int kh = 0; kh < 2; ++kh) {
      v16bf pf, bva, bvb;
#pragma unroll
      for (int e = 0; e < 16; ++e) {
        const int kk = kh * 32 + kmap(e, half);
        pf[e]  = Ps[wave][idx][kk];
        bva[e] = Vs[kk][idx];
        bvb[e] = Vs[kk][16 + idx];
      }
      oa = __builtin_amdgcn_wmma_f32_16x16x32_bf16(
          false, pf, false, bva, (short)0, oa, false, false);
      ob = __builtin_amdgcn_wmma_f32_16x16x32_bf16(
          false, pf, false, bvb, (short)0, ob, false, false);
    }
    __syncthreads();
  }

#pragma unroll
  for (int g = 0; g < 8; ++g) {
    const int q = qbase + wave * 16 + g + half * 8;
    if (q < LQ) {
      const float inv = rl[g] > 0.0f ? 1.0f / rl[g] : 0.0f;
      const size_t off = ((size_t)nb * LQ + q) * D_ + colOff;
      O[off + idx]      = oa[g] * inv;
      O[off + 16 + idx] = ob[g] * inv;
    }
  }
}

// ---------------------------------------------------------------------------
// LayerNorm with residual:  Y = LN(X + R) * g + b   (row length 256)
// ---------------------------------------------------------------------------
__global__ void __launch_bounds__(128)
layernorm_res(const float* __restrict__ X, const float* __restrict__ R,
              const float* __restrict__ g, const float* __restrict__ b,
              float* __restrict__ Y)
{
  __shared__ float red[128];
  const int row = blockIdx.x;
  const int t = threadIdx.x;
  const size_t base = (size_t)row * D_;
  float v0 = X[base + t]       + (R ? R[base + t]       : 0.0f);
  float v1 = X[base + t + 128] + (R ? R[base + t + 128] : 0.0f);
  red[t] = v0 + v1;
  __syncthreads();
  for (int o = 64; o > 0; o >>= 1) {
    if (t < o) red[t] += red[t + o];
    __syncthreads();
  }
  const float mu = red[0] * (1.0f / 256.0f);
  __syncthreads();
  const float d0 = v0 - mu, d1 = v1 - mu;
  red[t] = d0 * d0 + d1 * d1;
  __syncthreads();
  for (int o = 64; o > 0; o >>= 1) {
    if (t < o) red[t] += red[t + o];
    __syncthreads();
  }
  const float rstd = rsqrtf(red[0] * (1.0f / 256.0f) + 1e-5f);
  Y[base + t]       = d0 * rstd * g[t]       + b[t];
  Y[base + t + 128] = d1 * rstd * g[t + 128] + b[t + 128];
}

// ---------------------------------------------------------------------------
// Elementwise / small kernels
// ---------------------------------------------------------------------------

// X[row, j] + pe_table(pos = row % Lper)[j], D = 256
__global__ void pe_add(const float* __restrict__ X, float* __restrict__ Y,
                       int rows, int Lper)
{
  const int t = blockIdx.x * 256 + threadIdx.x;
  if (t >= rows * D_) return;
  const int j = t & 255;
  const int pos = (t >> 8) % Lper;
  const int ii = j & 127;
  const float ang = (float)pos * __expf(-(float)ii * (9.210340372f / 128.0f));
  const float pe = (j >= 128) ? __cosf(ang) : __sinf(ang);
  Y[t] = X[t] + pe;
}

// spatial_pe: pts strided (x,y) pairs -> [count, 256]
__global__ void spatial_pe_k(const float* __restrict__ pts, int stride,
                             int count, float* __restrict__ out)
{
  const int t = blockIdx.x * 256 + threadIdx.x;
  if (t >= count * D_) return;
  const int pt = t >> 8, j = t & 255;
  const float c = pts[(size_t)pt * stride + (j >= 128 ? 1 : 0)];
  const int jj = j & 127;
  const int ii = jj & 63;
  const float a = c * __expf(-(float)ii * (9.210340372f / 64.0f));
  out[t] = (jj >= 64) ? __cosf(a) : __sinf(a);
}

// scene[n,a,k,:2] = (proj[n,a] @ [anch_k, 1])[:2]
__global__ void scene_proj(const float* __restrict__ proj,
                           const float* __restrict__ anchors,
                           float* __restrict__ scene)
{
  const int t = blockIdx.x * 256 + threadIdx.x;
  if (t >= N_ * A_ * K_) return;
  const int k = t % K_;
  const int na = t / K_;
  const float ax = anchors[(size_t)k * T_ * 2 + (T_ - 1) * 2 + 0];
  const float ay = anchors[(size_t)k * T_ * 2 + (T_ - 1) * 2 + 1];
  const float* P = proj + (size_t)na * 9;
  scene[(size_t)t * 2 + 0] = P[0] * ax + P[1] * ay + P[2];
  scene[(size_t)t * 2 + 1] = P[3] * ax + P[4] * ay + P[5];
}

__global__ void build_queries(const float* __restrict__ ctx,
                              const float* __restrict__ embA,
                              const float* __restrict__ embS,
                              const float* __restrict__ embC,
                              const float* __restrict__ embG,
                              float* __restrict__ out)
{
  const int t = blockIdx.x * 256 + threadIdx.x;
  if (t >= N_ * LQ_ * D_) return;
  const int d = t & 255;
  const int q = (t >> 8) % LQ_;
  const int n = (t >> 8) / LQ_;
  const int a = q / K_, k = q % K_;
  const size_t sidx = (((size_t)(n * A_ + a)) * K_ + k) * D_ + d;
  out[t] = ctx[(size_t)q * D_ + d] + embA[(size_t)k * D_ + d] + embS[sidx] +
           embC[((size_t)(n * A_ + a)) * D_ + d] + embG[sidx];
}

// goal init: goal[n, kk*A + a] = scene[n, a, 0, :]   (jnp.tile semantics)
__global__ void goal_init(const float* __restrict__ scene,
                          float* __restrict__ goal)
{
  const int t = blockIdx.x * 256 + threadIdx.x;
  if (t >= N_ * LQ_) return;
  const int n = t / LQ_, q = t % LQ_;
  const int a = q % A_;
  const size_t s = (((size_t)(n * A_ + a)) * K_ + 0) * 2;
  goal[(size_t)t * 2 + 0] = scene[s + 0];
  goal[(size_t)t * 2 + 1] = scene[s + 1];
}

// goal[n,q] = (proj_k[n,q] @ [traj_end, 1])[:2], proj_k[n,q] = proj[n, q%A]
__global__ void goal_update(const float* __restrict__ traj,
                            const float* __restrict__ proj,
                            float* __restrict__ goal)
{
  const int t = blockIdx.x * 256 + threadIdx.x;
  if (t >= N_ * LQ_) return;
  const int n = t / LQ_, q = t % LQ_;
  const int a = q % A_;
  const float* tp = traj + ((size_t)t * T_ + (T_ - 1)) * 5;
  const float ex = tp[0], ey = tp[1];
  const float* P = proj + ((size_t)(n * A_ + a)) * 9;
  goal[(size_t)t * 2 + 0] = P[0] * ex + P[1] * ey + P[2];
  goal[(size_t)t * 2 + 1] = P[3] * ex + P[4] * ey + P[5];
}

// deformable attention: bilinear sample of bevval, weighted by softmaxed aw
__global__ void deform_sample(const float* __restrict__ bevval,
                              const float* __restrict__ doff,
                              const float* __restrict__ dattn,
                              const float* __restrict__ goal,
                              float* __restrict__ out)
{
  const int t = blockIdx.x * 256 + threadIdx.x;
  if (t >= N_ * LQ_ * D_) return;
  const int dd = t & 31;
  const int h = (t >> 5) & 7;
  const int row = t >> 8;               // n*LQ + q
  const int n = row / LQ_;

  const float gx = goal[(size_t)row * 2 + 0];
  const float gy = goal[(size_t)row * 2 + 1];
  const float rx = fminf(fmaxf((gx + 51.2f) * (1.0f / 102.4f), 0.0f), 1.0f);
  const float ry = fminf(fmaxf((gy + 51.2f) * (1.0f / 102.4f), 0.0f), 1.0f);

  // softmax of attention weights over P
  float aw[P_];
  float mx = -1e30f;
#pragma unroll
  for (int p = 0; p < P_; ++p) {
    aw[p] = dattn[(size_t)row * (H_ * P_) + h * P_ + p];
    mx = fmaxf(mx, aw[p]);
  }
  float sum = 0.0f;
#pragma unroll
  for (int p = 0; p < P_; ++p) { aw[p] = __expf(aw[p] - mx); sum += aw[p]; }
  const float inv = 1.0f / sum;

  const float* vb = bevval + (size_t)n * BEV_ * D_;
  const int ch = h * 32 + dd;
  float acc = 0.0f;
#pragma unroll
  for (int p = 0; p < P_; ++p) {
    const float ox = doff[(size_t)row * (H_ * P_ * 2) + (h * P_ + p) * 2 + 0];
    const float oy = doff[(size_t)row * (H_ * P_ * 2) + (h * P_ + p) * 2 + 1];
    const float px = (rx + ox * (1.0f / WB_)) * WB_ - 0.5f;
    const float py = (ry + oy * (1.0f / HB_)) * HB_ - 0.5f;
    const float x0 = floorf(px), y0 = floorf(py);
    const float wx = px - x0, wy = py - y0;
    float smp = 0.0f;
#pragma unroll
    for (int cy = 0; cy < 2; ++cy)
#pragma unroll
      for (int cx = 0; cx < 2; ++cx) {
        const float xi = x0 + cx, yi = y0 + cy;
        const bool valid = (xi >= 0.0f) && (xi < (float)WB_) &&
                           (yi >= 0.0f) && (yi < (float)HB_);
        const int xc = min(max((int)fmaxf(xi, 0.0f), 0), WB_ - 1);
        const int yc = min(max((int)fmaxf(yi, 0.0f), 0), HB_ - 1);
        float g = vb[((size_t)(yc * WB_ + xc)) * D_ + ch];
        g = valid ? g : 0.0f;
        const float w = (cx ? wx : 1.0f - wx) * (cy ? wy : 1.0f - wy);
        smp += g * w;
      }
    acc += aw[p] * inv * smp;
  }
  out[(size_t)row * D_ + ch] = acc;
}

__global__ void concat3(const float* __restrict__ a, const float* __restrict__ b,
                        const float* __restrict__ c, float* __restrict__ out,
                        int rows)
{
  const int t = blockIdx.x * 256 + threadIdx.x;
  if (t >= rows * 3 * D_) return;
  const int col = t % (3 * D_);
  const int row = t / (3 * D_);
  float v;
  if (col < D_)            v = a[(size_t)row * D_ + col];
  else if (col < 2 * D_)   v = b[(size_t)row * D_ + col - D_];
  else                     v = c[(size_t)row * D_ + col - 2 * D_];
  out[t] = v;
}

// softmax over K of score_out[n*LQ + a*K + k] -> scores[n, a, k]
__global__ void softmax_modes(const float* __restrict__ sc,
                              float* __restrict__ out)
{
  const int t = blockIdx.x * 256 + threadIdx.x;
  if (t >= N_ * A_) return;
  float v[K_], mx = -1e30f;
#pragma unroll
  for (int k = 0; k < K_; ++k) {
    v[k] = sc[(size_t)t * K_ + k];
    mx = fmaxf(mx, v[k]);
  }
  float sum = 0.0f;
#pragma unroll
  for (int k = 0; k < K_; ++k) { v[k] = __expf(v[k] - mx); sum += v[k]; }
  const float inv = 1.0f / sum;
#pragma unroll
  for (int k = 0; k < K_; ++k) out[(size_t)t * K_ + k] = v[k] * inv;
}

// ---------------------------------------------------------------------------
// Host orchestration
// ---------------------------------------------------------------------------
struct Lin { const float *b, *w; };
struct LNp { const float *b, *g; };
struct MHAp { Lin k, o, q, v; };
struct MLPp { Lin l1, l2; };
struct LayerP {
  MHAp agent; Lin dattn, doff, dout, dv; LNp ln1, ln2, ln3; MLPp mlp; MHAp self;
};

static inline void gemm(const float* A, const float* W, const float* b,
                        float* C, int M, int N, int K, int relu,
                        hipStream_t s)
{
  dim3 grid((N + 63) / 64, (M + 127) / 128);
  if (relu)
    gemm_bias_act<1><<<grid, 128, 0, s>>>(A, W, b, C, M, N, K);
  else
    gemm_bias_act<0><<<grid, 128, 0, s>>>(A, W, b, C, M, N, K);
}

static inline int blks(long n) { return (int)((n + 255) / 256); }

extern "C" void kernel_launch(void* const* d_in, const int* in_sizes, int n_in,
                              void* d_out, int out_size, void* d_ws,
                              size_t ws_size, hipStream_t stream)
{
  (void)in_sizes; (void)n_in; (void)out_size; (void)ws_size;

  // ---- unpack params (jax pytree order: dict keys sorted, lists in order;
  //      top-level inputs in setup_inputs() insertion order) ----
  int pi = 0;
  auto nextp = [&]() { return (const float*)d_in[pi++]; };
  auto rdlin = [&]() { Lin l; l.b = nextp(); l.w = nextp(); return l; };
  auto rdmlp = [&]() { MLPp m; m.l1 = rdlin(); m.l2 = rdlin(); return m; };
  auto rdmha = [&]() {
    MHAp m; m.k = rdlin(); m.o = rdlin(); m.q = rdlin(); m.v = rdlin();
    return m;
  };
  auto rdln = [&]() { LNp l; l.b = nextp(); l.g = nextp(); return l; };

  MLPp agent_query_mlp = rdmlp();
  Lin  anchor_fc = rdlin();
  const float* ctx_emb = nextp();
  Lin  cur_fc  = rdlin();
  Lin  goal_fc = rdlin();
  LayerP layers[L_];
  for (int li = 0; li < L_; ++li) {
    LayerP& L = layers[li];
    L.agent = rdmha();
    L.dattn = rdlin(); L.doff = rdlin(); L.dout = rdlin(); L.dv = rdlin();
    L.ln1 = rdln(); L.ln2 = rdln(); L.ln3 = rdln();
    L.mlp = rdmlp();
    L.self = rdmha();
  }
  MLPp map_query_mlp = rdmlp();
  Lin  scene_fc  = rdlin();
  MLPp score_mlp = rdmlp();
  MLPp traj_mlp  = rdmlp();

  const float* agent_current_pos = nextp();   // [N, A, 2]
  const float* agent_anchors     = nextp();   // [K, T, 2]
  const float* bev               = nextp();   // [N, 40000, 256]
  const float* agent_queries_in  = nextp();   // [N, A, 256]
  const float* map_queries_in    = nextp();   // [N, M, 256]
  const float* proj              = nextp();   // [N, A, 3, 3]

  // ---- workspace carve ----
  float* wsp = (float*)d_ws;
  size_t off = 0;
  auto alloc = [&](size_t n) { float* p = wsp + off; off += n; return p; };
  float* bevval  = alloc((size_t)N_ * BEV_ * D_);    // 20.48M
  float* queries = alloc((size_t)N_ * LQ_ * D_);
  float* sa      = alloc((size_t)N_ * LQ_ * D_);
  float* ag      = alloc((size_t)N_ * LQ_ * D_);
  float* mp      = alloc((size_t)N_ * LQ_ * D_);
  float* gp      = alloc((size_t)N_ * LQ_ * D_);
  float* qproj   = alloc((size_t)N_ * LQ_ * D_);
  float* kproj   = alloc((size_t)N_ * LQ_ * D_);
  float* vproj   = alloc((size_t)N_ * LQ_ * D_);
  float* attn    = alloc((size_t)N_ * LQ_ * D_);
  float* tmp     = alloc((size_t)N_ * LQ_ * D_);
  float* hid     = alloc((size_t)N_ * LQ_ * FF_);    // shared hidden buffer
  float* cat     = alloc((size_t)N_ * LQ_ * 3 * D_);
  float* aq      = alloc((size_t)N_ * A_ * D_);
  float* mq      = alloc((size_t)N_ * M_ * D_);
  float* aq_in   = alloc((size_t)N_ * A_ * D_);
  float* mq_in   = alloc((size_t)N_ * M_ * D_);
  float* peA     = alloc((size_t)K_ * D_);
  float* peS     = alloc((size_t)N_ * A_ * K_ * D_);
  float* peC     = alloc((size_t)N_ * A_ * D_);
  float* embA    = alloc((size_t)K_ * D_);
  float* embS    = alloc((size_t)N_ * A_ * K_ * D_);
  float* embC    = alloc((size_t)N_ * A_ * D_);
  float* embG    = alloc((size_t)N_ * A_ * K_ * D_);
  float* scene   = alloc((size_t)N_ * A_ * K_ * 2);
  float* goal    = alloc((size_t)N_ * LQ_ * 2);
  float* doffO   = alloc((size_t)N_ * LQ_ * H_ * P_ * 2);
  float* dattnO  = alloc((size_t)N_ * LQ_ * H_ * P_);
  float* scoreO  = alloc((size_t)N_ * LQ_);

  float* out_mq    = (float*)d_out;                       // [N, A, K, D]
  float* out_traj  = out_mq + (size_t)N_ * LQ_ * D_;      // [N, LQ, T, 5]
  float* out_score = out_traj + (size_t)N_ * LQ_ * T_ * 5;

  const float attn_scale = 0.17677669529663687f;  // 1/sqrt(32)

  // ---- agent / map query MLPs with sinusoidal PE ----
  pe_add<<<blks((long)N_ * A_ * D_), 256, 0, stream>>>(agent_queries_in, aq_in,
                                                       N_ * A_, A_);
  gemm(aq_in, agent_query_mlp.l1.w, agent_query_mlp.l1.b, hid,
       N_ * A_, FF_, D_, 1, stream);
  gemm(hid, agent_query_mlp.l2.w, agent_query_mlp.l2.b, aq,
       N_ * A_, D_, FF_, 0, stream);

  pe_add<<<blks((long)N_ * M_ * D_), 256, 0, stream>>>(map_queries_in, mq_in,
                                                       N_ * M_, M_);
  gemm(mq_in, map_query_mlp.l1.w, map_query_mlp.l1.b, hid,
       N_ * M_, FF_, D_, 1, stream);
  gemm(hid, map_query_mlp.l2.w, map_query_mlp.l2.b, mq,
       N_ * M_, D_, FF_, 0, stream);

  // ---- anchors -> scene / spatial embeddings -> initial queries ----
  scene_proj<<<blks((long)N_ * A_ * K_), 256, 0, stream>>>(proj, agent_anchors,
                                                           scene);
  spatial_pe_k<<<blks((long)K_ * D_), 256, 0, stream>>>(
      agent_anchors + (T_ - 1) * 2, T_ * 2, K_, peA);
  spatial_pe_k<<<blks((long)N_ * A_ * K_ * D_), 256, 0, stream>>>(
      scene, 2, N_ * A_ * K_, peS);
  spatial_pe_k<<<blks((long)N_ * A_ * D_), 256, 0, stream>>>(
      agent_current_pos, 2, N_ * A_, peC);
  gemm(peA, anchor_fc.w, anchor_fc.b, embA, K_, D_, D_, 0, stream);
  gemm(peS, scene_fc.w, scene_fc.b, embS, N_ * A_ * K_, D_, D_, 0, stream);
  gemm(peC, cur_fc.w, cur_fc.b, embC, N_ * A_, D_, D_, 0, stream);
  gemm(peS, goal_fc.w, goal_fc.b, embG, N_ * A_ * K_, D_, D_, 0, stream);
  build_queries<<<blks((long)N_ * LQ_ * D_), 256, 0, stream>>>(
      ctx_emb, embA, embS, embC, embG, queries);
  goal_init<<<blks((long)N_ * LQ_), 256, 0, stream>>>(scene, goal);

  // ---- decoder layers ----
  const int rowsQ = N_ * LQ_;
  for (int li = 0; li < L_; ++li) {
    const LayerP& L = layers[li];

    // self attention
    gemm(queries, L.self.q.w, L.self.q.b, qproj, rowsQ, D_, D_, 0, stream);
    gemm(queries, L.self.k.w, L.self.k.b, kproj, rowsQ, D_, D_, 0, stream);
    gemm(queries, L.self.v.w, L.self.v.b, vproj, rowsQ, D_, D_, 0, stream);
    flash_attn<<<dim3((LQ_ + 63) / 64, H_, N_), 128, 0, stream>>>(
        qproj, kproj, vproj, attn, LQ_, LQ_, attn_scale);
    gemm(attn, L.self.o.w, L.self.o.b, tmp, rowsQ, D_, D_, 0, stream);
    layernorm_res<<<rowsQ, 128, 0, stream>>>(tmp, queries, L.ln1.g, L.ln1.b, sa);

    // agent cross attention
    gemm(sa, L.agent.q.w, L.agent.q.b, qproj, rowsQ, D_, D_, 0, stream);
    gemm(aq, L.agent.k.w, L.agent.k.b, kproj, N_ * A_, D_, D_, 0, stream);
    gemm(aq, L.agent.v.w, L.agent.v.b, vproj, N_ * A_, D_, D_, 0, stream);
    flash_attn<<<dim3((LQ_ + 63) / 64, H_, N_), 128, 0, stream>>>(
        qproj, kproj, vproj, attn, LQ_, A_, attn_scale);
    gemm(attn, L.agent.o.w, L.agent.o.b, tmp, rowsQ, D_, D_, 0, stream);
    layernorm_res<<<rowsQ, 128, 0, stream>>>(tmp, sa, L.ln2.g, L.ln2.b, ag);

    // map cross attention (same q projection weights -> reuse qproj)
    gemm(mq, L.agent.k.w, L.agent.k.b, kproj, N_ * M_, D_, D_, 0, stream);
    gemm(mq, L.agent.v.w, L.agent.v.b, vproj, N_ * M_, D_, D_, 0, stream);
    flash_attn<<<dim3((LQ_ + 63) / 64, H_, N_), 128, 0, stream>>>(
        qproj, kproj, vproj, attn, LQ_, M_, attn_scale);
    gemm(attn, L.agent.o.w, L.agent.o.b, tmp, rowsQ, D_, D_, 0, stream);
    layernorm_res<<<rowsQ, 128, 0, stream>>>(tmp, sa, L.ln3.g, L.ln3.b, mp);

    // deformable attention over BEV
    gemm(bev, L.dv.w, L.dv.b, bevval, N_ * BEV_, D_, D_, 0, stream);
    gemm(queries, L.doff.w, L.doff.b, doffO, rowsQ, H_ * P_ * 2, D_, 0, stream);
    gemm(queries, L.dattn.w, L.dattn.b, dattnO, rowsQ, H_ * P_, D_, 0, stream);
    deform_sample<<<blks((long)rowsQ * D_), 256, 0, stream>>>(
        bevval, doffO, dattnO, goal, qproj);  // qproj reused as sample buffer
    gemm(qproj, L.dout.w, L.dout.b, gp, rowsQ, D_, D_, 0, stream);

    // concat + MLP -> new queries
    concat3<<<blks((long)rowsQ * 3 * D_), 256, 0, stream>>>(ag, mp, gp, cat,
                                                            rowsQ);
    gemm(cat, L.mlp.l1.w, L.mlp.l1.b, hid, rowsQ, FF_, 3 * D_, 1, stream);
    gemm(hid, L.mlp.l2.w, L.mlp.l2.b, queries, rowsQ, D_, FF_, 0, stream);

    // trajectory head (last layer's output remains in d_out)
    gemm(queries, traj_mlp.l1.w, traj_mlp.l1.b, hid, rowsQ, FF_, D_, 1, stream);
    gemm(hid, traj_mlp.l2.w, traj_mlp.l2.b, out_traj, rowsQ, T_ * 5, FF_, 0,
         stream);
    goal_update<<<blks((long)N_ * LQ_), 256, 0, stream>>>(out_traj, proj, goal);
  }

  // ---- outputs: motion queries, scores ----
  hipMemcpyAsync(out_mq, queries, (size_t)N_ * LQ_ * D_ * sizeof(float),
                 hipMemcpyDeviceToDevice, stream);
  gemm(queries, score_mlp.l1.w, score_mlp.l1.b, hid, rowsQ, FF_, D_, 1, stream);
  gemm(hid, score_mlp.l2.w, score_mlp.l2.b, scoreO, rowsQ, 1, FF_, 0, stream);
  softmax_modes<<<blks((long)N_ * A_), 256, 0, stream>>>(scoreO, out_score);
}
